// HardLayer_14061722927181
// MI455X (gfx1250) — matrix operand order repaired and checked by reading, not compiled, not run on hardware
//
#include <hip/hip_runtime.h>
#include <hip/hip_bf16.h>
#include <stdint.h>

// LSTM (hard-sigmoid/hard-tanh) for MI455X / gfx1250, wave32 + WMMA bf16.
// T=512, B=32, I=H=1024.  out = y_all[512,32,1024] || c_all[512,32,1024] fp32.

typedef __attribute__((ext_vector_type(16))) __bf16 v16b;
typedef __attribute__((ext_vector_type(8)))  float  v8f;

#define T_STEPS 512
#define BATCH   32
#define IDIM    1024
#define HDIM    1024
#define GDIM    4096          // 4*H
#define NBLK    64            // persistent blocks, one per 16 h-columns

// Workspace layout (bytes). Needs ~160 MiB.
#define WS_WB    ((size_t)0)            // W bf16: 4096*1024*2 = 8 MiB
#define WS_RB    ((size_t)8u  << 20)    // R bf16: 8 MiB
#define WS_Y     ((size_t)16u << 20)    // y ping-pong: 2*32*1024*2 = 128 KiB
#define WS_CNT   (WS_Y + (1u << 18))    // grid-barrier counter
#define WS_GATES ((size_t)32u << 20)    // gates_pre bf16: 512*32*4096*2 = 128 MiB

#define SR_STRIDE 1032                  // 1024 + 8 bf16 pad (LDS bank spread)

__device__ __forceinline__ float hsig(float x)  { return __saturatef(x * (1.0f/6.0f) + 0.5f); }
__device__ __forceinline__ float htanh(float x) { return fminf(fmaxf(x, -1.0f), 1.0f); }

__global__ void cvt_f32_bf16(const float* __restrict__ s, __bf16* __restrict__ d, int n) {
    int i = blockIdx.x * blockDim.x + threadIdx.x;
    int stride = gridDim.x * blockDim.x;
    for (; i < n; i += stride) d[i] = (__bf16)s[i];
}

__global__ void init_kernel(const float* __restrict__ y0, __bf16* __restrict__ yws,
                            unsigned* __restrict__ cnt) {
    if (blockIdx.x == 0 && threadIdx.x == 0) *cnt = 0u;
    int i = blockIdx.x * blockDim.x + threadIdx.x;
    int stride = gridDim.x * blockDim.x;
    for (; i < BATCH * HDIM; i += stride) yws[i] = (__bf16)y0[i];
}

// ---------------------------------------------------------------------------
// Phase 1: gates_pre[t*B+b, g] = x[t,b,:] . W[g,:] + bW[g] + bR[g]   (bf16 out)
// GEMM M=16384 N=4096 K=1024.  One wave -> 16x64 strip (A frag reused 4x).
// ---------------------------------------------------------------------------
__global__ void __launch_bounds__(256)
gates_kernel(const float* __restrict__ x, const __bf16* __restrict__ Wb,
             const float* __restrict__ bW, const float* __restrict__ bR,
             __bf16* __restrict__ gates) {
    const int lane   = threadIdx.x & 31;
    const int wave   = blockIdx.x * (blockDim.x >> 5) + (threadIdx.x >> 5);
    const int NSTRIP = GDIM / 64;               // 64
    const int mtile  = wave / NSTRIP;           // 0..1023
    const int nstrip = wave % NSTRIP;
    const int ln15   = lane & 15;
    const int hi     = lane >> 4;               // K-half select
    const int arow   = mtile * 16 + ln15;       // flattened (t*B+b)

    v8f acc[4] = {};
    const float* xrow = x + (size_t)arow * IDIM;

    for (int kb = 0; kb < IDIM; kb += 32) {
        // A fragment: rows in lanes, K chunks {c0..c0+7, c0+16..c0+23}, c0 = kb + hi*8
        const int c0 = kb + hi * 8;
        const float4 f0 = *(const float4*)(xrow + c0);
        const float4 f1 = *(const float4*)(xrow + c0 + 4);
        const float4 f2 = *(const float4*)(xrow + c0 + 16);
        const float4 f3 = *(const float4*)(xrow + c0 + 20);
        v16b a;
        a[0]=(__bf16)f0.x; a[1]=(__bf16)f0.y; a[2]=(__bf16)f0.z; a[3]=(__bf16)f0.w;
        a[4]=(__bf16)f1.x; a[5]=(__bf16)f1.y; a[6]=(__bf16)f1.z; a[7]=(__bf16)f1.w;
        a[8]=(__bf16)f2.x; a[9]=(__bf16)f2.y; a[10]=(__bf16)f2.z; a[11]=(__bf16)f2.w;
        a[12]=(__bf16)f3.x; a[13]=(__bf16)f3.y; a[14]=(__bf16)f3.z; a[15]=(__bf16)f3.w;
        #pragma unroll
        for (int j = 0; j < 4; ++j) {
            // B fragment: lane = column, K chunks {c0..c0+7, c0+16..c0+23}, c0 = kb + hi*8
            const int ncol = nstrip * 64 + j * 16 + ln15;
            const __bf16* bp = Wb + (size_t)ncol * IDIM + kb + hi * 8;
            v16b b;
            *(uint4*)&b       = *(const uint4*)bp;
            *((uint4*)&b + 1) = *(const uint4*)(bp + 16);
            acc[j] = __builtin_amdgcn_wmma_f32_16x16x32_bf16(
                false, a, false, b, (short)0, acc[j], false, false);
        }
    }
    #pragma unroll
    for (int j = 0; j < 4; ++j) {
        const int ncol = nstrip * 64 + j * 16 + ln15;   // column fixed per lane
        const float bias = bW[ncol] + bR[ncol];
        #pragma unroll
        for (int r = 0; r < 8; ++r) {                    // C layout: M = r + 8*hi
            const int m = mtile * 16 + hi * 8 + r;
            gates[(size_t)m * GDIM + ncol] = (__bf16)(acc[j][r] + bias);
        }
    }
}

// ---------------------------------------------------------------------------
// Phase 2: persistent recurrent kernel. 64 blocks x 256 threads (8 waves).
// Wave w: gate = w&3, mtile = w>>2. Block bh owns h-columns [bh*16, bh*16+16).
// R slice (4 gates x 16 cols x 1024 K bf16 = 128 KiB) staged in LDS once.
// c state lives in registers for all 512 steps.
// ---------------------------------------------------------------------------
__global__ void __launch_bounds__(256, 1)
lstm_kernel(const float* __restrict__ c0, const __bf16* __restrict__ Rb,
            const __bf16* __restrict__ gates, __bf16* __restrict__ yws,
            unsigned* __restrict__ cnt, float* __restrict__ y_all,
            float* __restrict__ c_all) {
    extern __shared__ char smem[];
    __bf16* sR = (__bf16*)smem;                               // [64][SR_STRIDE]
    float*  sG = (float*)(smem + 64 * SR_STRIDE * 2);         // [4][32][16]

    const int tid   = threadIdx.x;
    const int lane  = tid & 31;
    const int w     = tid >> 5;
    const int gate  = w & 3;
    const int mtile = w >> 2;
    const int bh    = blockIdx.x;
    const int ln15  = lane & 15;
    const int hi    = lane >> 4;

    // Stage R slice: local col c -> global col (c>>4)*HDIM + bh*16 + (c&15)
    for (int idx = tid; idx < 64 * 128; idx += 256) {         // 128 x uint4 per col
        const int c = idx >> 7, q = idx & 127;
        const int gc = (c >> 4) * HDIM + bh * 16 + (c & 15);
        *(uint4*)(sR + c * SR_STRIDE + q * 8) =
            *(const uint4*)(Rb + (size_t)gc * HDIM + q * 8);
    }

    // c state: 2 (b,h) pairs per thread, registers for the whole scan
    float creg[2];
    int pb[2], ph[2];
    #pragma unroll
    for (int p = 0; p < 2; ++p) {
        const int pp = tid + p * 256;
        pb[p] = pp >> 4; ph[p] = pp & 15;
        creg[p] = c0[(size_t)pb[p] * HDIM + bh * 16 + ph[p]];
    }
    __syncthreads();

    const int arow = mtile * 16 + ln15;
    const __bf16* sB0 = sR + (gate * 16 + ln15) * SR_STRIDE + hi * 8;
    // this wave's gates_pre base (column fixed per lane, 8 rows at stride GDIM)
    const __bf16* gbase = gates + ((size_t)(mtile * 16 + hi * 8)) * GDIM
                                + gate * HDIM + bh * 16 + ln15;

    for (int t = 0; t < T_STEPS; ++t) {
        // Issue this step's gates_pre loads FIRST so they overlap the GEMM.
        const __bf16* gp = gbase + (size_t)t * BATCH * GDIM;
        float gpre[8];
        #pragma unroll
        for (int r = 0; r < 8; ++r) gpre[r] = (float)gp[(size_t)r * GDIM];

        // Prefetch next step's gate slice into near caches (global_prefetch_b8).
        if (t + 1 < T_STEPS) {
            const __bf16* gpn = gp + (size_t)BATCH * GDIM;
            #pragma unroll
            for (int r = 0; r < 8; ++r)
                __builtin_prefetch((const void*)(gpn + (size_t)r * GDIM), 0, 0);
        }

        const __bf16* yprev = yws + (size_t)(t & 1) * (BATCH * HDIM) + (size_t)arow * HDIM;
        v8f acc = {};
        #pragma unroll 4
        for (int kb = 0; kb < HDIM; kb += 32) {
            v16b a, b;
            *(uint4*)&a       = *(const uint4*)(yprev + kb + hi * 8);        // K c0..c0+7
            *((uint4*)&a + 1) = *(const uint4*)(yprev + kb + hi * 8 + 16);   // K c0+16..+23
            *(uint4*)&b       = *(const uint4*)(sB0 + kb);                   // LDS, ds_load_b128
            *((uint4*)&b + 1) = *(const uint4*)(sB0 + kb + 16);
            acc = __builtin_amdgcn_wmma_f32_16x16x32_bf16(
                false, a, false, b, (short)0, acc, false, false);
        }

        // add gates_pre, exchange through LDS
        #pragma unroll
        for (int r = 0; r < 8; ++r) {
            const int b_ = mtile * 16 + hi * 8 + r;
            sG[(gate * 32 + b_) * 16 + ln15] = acc[r] + gpre[r];
        }
        __syncthreads();

        #pragma unroll
        for (int p = 0; p < 2; ++p) {
            const int b_ = pb[p], h_ = ph[p];
            const float iv = hsig (sG[(0 * 32 + b_) * 16 + h_]);
            const float fv = hsig (sG[(1 * 32 + b_) * 16 + h_]);
            const float zv = htanh(sG[(2 * 32 + b_) * 16 + h_]);
            const float ov = hsig (sG[(3 * 32 + b_) * 16 + h_]);
            const float cn = fv * creg[p] + iv * zv;
            const float yn = ov * htanh(cn);
            creg[p] = cn;
            const size_t oidx = ((size_t)t * BATCH + b_) * HDIM + bh * 16 + h_;
            y_all[oidx] = yn;
            c_all[oidx] = cn;
            yws[(size_t)((t + 1) & 1) * (BATCH * HDIM) + (size_t)b_ * HDIM + bh * 16 + h_] =
                (__bf16)yn;
        }

        // grid barrier (monotonic counter; 64 co-resident blocks)
        __syncthreads();
        if (tid == 0) {
            __threadfence();                       // release y writes to device scope
            atomicAdd(cnt, 1u);
            const unsigned target = (unsigned)NBLK * (unsigned)(t + 1);
            while (__hip_atomic_load(cnt, __ATOMIC_ACQUIRE, __HIP_MEMORY_SCOPE_AGENT) < target) {
                __builtin_amdgcn_s_sleep(2);
            }
        }
        __syncthreads();
        __threadfence();                           // acquire: invalidate stale y in near caches
    }
}

extern "C" void kernel_launch(void* const* d_in, const int* in_sizes, int n_in,
                              void* d_out, int out_size, void* d_ws, size_t ws_size,
                              hipStream_t stream) {
    const float* y0 = (const float*)d_in[0];
    const float* c0 = (const float*)d_in[1];
    const float* x  = (const float*)d_in[2];
    const float* W  = (const float*)d_in[3];
    const float* R  = (const float*)d_in[4];
    const float* bW = (const float*)d_in[5];
    const float* bR = (const float*)d_in[6];

    char* ws = (char*)d_ws;
    __bf16*   Wb    = (__bf16*)(ws + WS_WB);
    __bf16*   Rb    = (__bf16*)(ws + WS_RB);
    __bf16*   yws   = (__bf16*)(ws + WS_Y);
    unsigned* cnt   = (unsigned*)(ws + WS_CNT);
    __bf16*   gates = (__bf16*)(ws + WS_GATES);

    float* y_all = (float*)d_out;
    float* c_all = y_all + (size_t)T_STEPS * BATCH * HDIM;

    cvt_f32_bf16<<<512, 256, 0, stream>>>(W, Wb, GDIM * IDIM);
    cvt_f32_bf16<<<512, 256, 0, stream>>>(R, Rb, GDIM * HDIM);
    init_kernel <<<32, 256, 0, stream>>>(y0, yws, cnt);

    // Phase 1: 1024 m-tiles x 64 n-strips = 65536 waves, 8 waves/block
    gates_kernel<<<8192, 256, 0, stream>>>(x, Wb, bW, bR, gates);

    // Phase 2: persistent scan; >64 KiB dynamic LDS needs explicit opt-in
    const size_t shbytes = (size_t)64 * SR_STRIDE * 2 + (size_t)4 * 32 * 16 * 4; // ~140 KiB
    hipFuncSetAttribute((const void*)lstm_kernel,
                        hipFuncAttributeMaxDynamicSharedMemorySize, (int)shbytes);
    lstm_kernel<<<NBLK, 256, shbytes, stream>>>(c0, Rb, gates, yws, cnt, y_all, c_all);
}
